// MixTreeLSTMCell_39170101739917
// MI455X (gfx1250) — matrix operand order repaired and checked by reading, not compiled, not run on hardware
//
#include <hip/hip_runtime.h>
#include <hip/hip_bf16.h>
#include <math.h>

// ---------------------------------------------------------------------------
// Fused MixTreeLSTMCell for MI455X (gfx1250, wave32, WMMA bf16)
//
// Single-GEMM formulation:  A[N,832](bf16) x BT[2560,832](bf16, row = output
// column of the fused weight matrix), f32 accumulate via
// v_wmma_f32_16x16x32_bf16, epilogue fused through a 160KB LDS tile.
// ---------------------------------------------------------------------------

#define XS   300
#define HS   256
#define KP   832          // 300 + 512 padded up to a multiple of 32
#define COLS 2560         // 768 iou_n | 768 iou_s | 512 f_n | 256 f_s0 | 256 f_s1
#define MT   16           // nodes per workgroup (one 16-row WMMA M tile)
#define WAVES 16
#define TPW  10           // 160 column tiles / 16 waves

typedef __attribute__((ext_vector_type(16))) __bf16 v16bf;
typedef __attribute__((ext_vector_type(8)))  float  v8f;

union FragU { v16bf v; uint4 u[2]; };

// ---------------------------------------------------------------------------
// Kernel 0: pack activations A[node][k] = [x | h_child0 | h_child1 | 0-pad] bf16
// ---------------------------------------------------------------------------
__global__ void pack_a(const float* __restrict__ x,
                       const float* __restrict__ h_child,
                       __bf16* __restrict__ A, int n_nodes) {
  long idx = (long)blockIdx.x * blockDim.x + threadIdx.x;
  long total = (long)n_nodes * KP;
  if (idx >= total) return;
  int node = (int)(idx / KP);
  int k    = (int)(idx % KP);
  float v = 0.0f;
  if (k < XS)                v = x[(long)node * XS + k];
  else if (k < XS + 2 * HS)  v = h_child[(long)node * 2 * HS + (k - XS)];
  A[idx] = (__bf16)v;
}

// ---------------------------------------------------------------------------
// Kernel 1: pack fused weight matrix (transposed: BT[n][k]) + bias[n]
//   n <  768 : iou_n   row = [W_iou[n] | U_iou[n]]                 bias b_iou
//   n < 1536 : iou_s   row = [W_iou_s  | U_iou_s | U_iou_s]        bias b_iou_s
//   n < 2048 : f_n     row = [0        | U_f_w ]                   bias U_f_b
//   n < 2304 : f_s ch0 row = [0 | U_f_s_w | 0]                     bias U_f_s_b
//   n < 2560 : f_s ch1 row = [0 | 0 | U_f_s_w]                     bias U_f_s_b
// ---------------------------------------------------------------------------
__global__ void pack_b(const float* __restrict__ W_iou,  const float* __restrict__ U_iou,
                       const float* __restrict__ b_iou,  const float* __restrict__ U_f_w,
                       const float* __restrict__ U_f_b,  const float* __restrict__ W_iou_s,
                       const float* __restrict__ U_iou_s,const float* __restrict__ b_iou_s,
                       const float* __restrict__ U_f_s_w,const float* __restrict__ U_f_s_b,
                       __bf16* __restrict__ BT, float* __restrict__ bias) {
  const int n = blockIdx.x;
  for (int k = threadIdx.x; k < KP; k += blockDim.x) {
    float v = 0.0f;
    if (n < 768) {
      if (k < 300)       v = W_iou[n * 300 + k];
      else if (k < 812)  v = U_iou[n * 512 + (k - 300)];
    } else if (n < 1536) {
      int j = n - 768;
      if (k < 300)       v = W_iou_s[j * 300 + k];
      else if (k < 556)  v = U_iou_s[j * 256 + (k - 300)];
      else if (k < 812)  v = U_iou_s[j * 256 + (k - 556)];   // child-sum duplicate
    } else if (n < 2048) {
      int j = n - 1536;
      if (k >= 300 && k < 812) v = U_f_w[j * 512 + (k - 300)];
    } else if (n < 2304) {
      int j = n - 2048;
      if (k >= 300 && k < 556) v = U_f_s_w[j * 256 + (k - 300)];
    } else {
      int j = n - 2304;
      if (k >= 556 && k < 812) v = U_f_s_w[j * 256 + (k - 556)];
    }
    BT[(long)n * KP + k] = (__bf16)v;
  }
  if (threadIdx.x == 0) {
    float bv;
    if (n < 768)       bv = b_iou[n];
    else if (n < 1536) bv = b_iou_s[n - 768];
    else if (n < 2048) bv = U_f_b[n - 1536];
    else if (n < 2304) bv = U_f_s_b[n - 2048];
    else               bv = U_f_s_b[n - 2304];
    bias[n] = bv;
  }
}

// ---------------------------------------------------------------------------
// Kernel 2: fused GEMM (bf16 WMMA) + TreeLSTM gate epilogue.
// 512 threads = 16 waves; each wave: 10 column tiles, 26 K-steps.
// ---------------------------------------------------------------------------
__device__ __forceinline__ float sigm(float v) {
  return 1.0f / (1.0f + __expf(-v));
}

__global__ __launch_bounds__(512)
void treelstm_wmma(const __bf16* __restrict__ A, const __bf16* __restrict__ BT,
                   const float* __restrict__ bias,
                   const float* __restrict__ c_child, const int* __restrict__ tvec,
                   float* __restrict__ out, int n_nodes) {
  __shared__ float smem[MT * COLS];          // 160 KB (< 320 KB WGP LDS)

  const int tid   = threadIdx.x;
  const int wave  = tid >> 5;
  const int lane  = tid & 31;
  const int lhalf = lane >> 4;               // 0: lanes 0-15, 1: lanes 16-31
  const int l15   = lane & 15;
  const long node0 = (long)blockIdx.x * MT;

  v8f acc[TPW];
  #pragma unroll
  for (int i = 0; i < TPW; ++i) acc[i] = (v8f){0,0,0,0,0,0,0,0};

  // A fragment (16x32 bf16): lane row = l15; lanes 0-15 hold K {0..7,16..23},
  // lanes 16-31 hold K {8..15,24..31}  -> two contiguous 16B loads per step.
  const __bf16* Arow = A + (node0 + l15) * KP + (lhalf ? 8 : 0);

  for (int kk = 0; kk < KP / 32; ++kk) {
    FragU a;
    a.u[0] = *(const uint4*)(Arow + kk * 32);
    a.u[1] = *(const uint4*)(Arow + kk * 32 + 16);
    #pragma unroll
    for (int tI = 0; tI < TPW; ++tI) {
      // B fragment (32x16 bf16): lane col = l15; lanes 0-15 hold K {0..7,16..23},
      // lanes 16-31 hold K {8..15,24..31} -> two contiguous 16B loads from BT row.
      const int n = (wave * TPW + tI) * 16 + l15;
      const __bf16* Brow = BT + (long)n * KP + kk * 32 + lhalf * 8;
      FragU b;
      b.u[0] = *(const uint4*)(Brow);
      b.u[1] = *(const uint4*)(Brow + 16);
      acc[tI] = __builtin_amdgcn_wmma_f32_16x16x32_bf16(
          false, a.v, false, b.v, (short)0, acc[tI], false, false);
    }
  }

  // C tile (16x16 f32): VGPR r, lanes 0-15 -> M=r, lanes 16-31 -> M=r+8; N=l15.
  #pragma unroll
  for (int tI = 0; tI < TPW; ++tI) {
    const int n  = (wave * TPW + tI) * 16 + l15;
    const float bn = bias[n];
    #pragma unroll
    for (int r = 0; r < 8; ++r) {
      const int m = r + (lhalf << 3);
      smem[m * COLS + n] = acc[tI][r] + bn;
    }
  }
  __syncthreads();

  // Epilogue: 16 nodes x 256 features = 4096 elems, 8 per thread (uniform).
  for (int p = tid; p < MT * HS; p += 512) {
    const int m = p >> 8;
    const int j = p & 255;
    const float* L = smem + m * COLS;
    const long node = node0 + m;
    const float cc0 = c_child[node * 2 * HS + j];
    const float cc1 = c_child[node * 2 * HS + HS + j];
    const int   ty  = tvec[node];

    const float i_n = L[j],        o_n = L[256 + j],  u_n = L[512 + j];
    const float i_s = L[768 + j],  o_s = L[1024 + j], u_s = L[1280 + j];
    const float fn0 = L[1536 + j], fn1 = L[1792 + j];
    const float fs0 = L[2048 + j], fs1 = L[2304 + j];

    const float c_n = sigm(fn0) * cc0 + sigm(fn1) * cc1;
    const float c_s = sigm(fs0) * cc0 + sigm(fs1) * cc1;

    const bool s = ty > 0;
    const float iv = s ? i_s : i_n;
    const float ov = s ? o_s : o_n;
    const float uv = s ? u_s : u_n;
    const float cr = s ? c_s : c_n;

    const float c = sigm(iv) * tanhf(uv) + cr;
    const float h = sigm(ov) * tanhf(c);
    out[node * HS + j] = h;
    out[(long)n_nodes * HS + node * HS + j] = c;
  }
}

// ---------------------------------------------------------------------------
// Host launcher
// ---------------------------------------------------------------------------
extern "C" void kernel_launch(void* const* d_in, const int* in_sizes, int n_in,
                              void* d_out, int out_size, void* d_ws, size_t ws_size,
                              hipStream_t stream) {
  const float* x        = (const float*)d_in[0];
  const float* h_child  = (const float*)d_in[1];
  const float* c_child  = (const float*)d_in[2];
  const int*   tvec     = (const int*)d_in[3];
  const float* W_iou    = (const float*)d_in[4];
  const float* U_iou    = (const float*)d_in[5];
  const float* b_iou    = (const float*)d_in[6];
  const float* U_f_w    = (const float*)d_in[7];
  const float* U_f_b    = (const float*)d_in[8];
  const float* W_iou_s  = (const float*)d_in[9];
  const float* U_iou_s  = (const float*)d_in[10];
  const float* b_iou_s  = (const float*)d_in[11];
  const float* U_f_s_w  = (const float*)d_in[12];
  const float* U_f_s_b  = (const float*)d_in[13];

  const int n_nodes = in_sizes[0] / XS;

  // Workspace carve-out: A (bf16 N x 832), BT (bf16 2560 x 832), bias (f32 2560)
  char* ws = (char*)d_ws;
  size_t a_bytes = ((size_t)n_nodes * KP * 2 + 255) & ~(size_t)255;
  size_t b_bytes = ((size_t)COLS * KP * 2 + 255) & ~(size_t)255;
  __bf16* Abuf = (__bf16*)ws;
  __bf16* BT   = (__bf16*)(ws + a_bytes);
  float*  bias = (float*)(ws + a_bytes + b_bytes);

  const long totalA = (long)n_nodes * KP;
  pack_a<<<(int)((totalA + 255) / 256), 256, 0, stream>>>(x, h_child, Abuf, n_nodes);
  pack_b<<<COLS, 256, 0, stream>>>(W_iou, U_iou, b_iou, U_f_w, U_f_b,
                                   W_iou_s, U_iou_s, b_iou_s, U_f_s_w, U_f_s_b,
                                   BT, bias);
  treelstm_wmma<<<n_nodes / MT, 512, 0, stream>>>(Abuf, BT, bias, c_child, tvec,
                                                  (float*)d_out, n_nodes);
}
